// GHConv_52613349376205
// MI455X (gfx1250) — matrix operand order repaired and checked by reading, not compiled, not run on hardware
//
#include <hip/hip_runtime.h>
#include <stdint.h>

// ---------------------------------------------------------------------------
// GHConv on MI455X (gfx1250, wave32, WMMA bf16 path)
// ---------------------------------------------------------------------------

typedef __attribute__((ext_vector_type(16))) __bf16          v16bf;
typedef __attribute__((ext_vector_type(16))) unsigned short  v16u;
typedef __attribute__((ext_vector_type(8)))  unsigned short  v8u;
typedef __attribute__((ext_vector_type(8)))  float           v8f;

__device__ __forceinline__ unsigned short f32_to_bf16(float f) {
    unsigned int u = __builtin_bit_cast(unsigned int, f);
    unsigned int r = u + 0x7FFFu + ((u >> 16) & 1u);   // round-to-nearest-even
    return (unsigned short)(r >> 16);
}

__device__ __forceinline__ uint32_t pack_bf16x2(float lo, float hi) {
    return (uint32_t)f32_to_bf16(lo) | ((uint32_t)f32_to_bf16(hi) << 16);
}

__device__ __forceinline__ v16bf concat_frag(v8u lo, v8u hi) {
    v16u u = __builtin_shufflevector(lo, hi, 0, 1, 2, 3, 4, 5, 6, 7,
                                             8, 9, 10, 11, 12, 13, 14, 15);
    return __builtin_bit_cast(v16bf, u);
}

// A fragment: 16x32 bf16 (M x K). LDS tile M-major, row stride 40 ushorts
// (80B: 16B-aligned chunks, 16-bank spread). Lane (m=L%16, half=L/16) holds
// K = {8h..8h+7} and {16+8h..+7}: two 16B chunks -> 2x ds_load_b128.
__device__ __forceinline__ v16bf load_a_frag(const unsigned short* as, int lda,
                                             int m0, int lane) {
    const unsigned short* row = as + (m0 + (lane & 15)) * lda + 8 * (lane >> 4);
    v8u lo = *(const v8u*)row;
    v8u hi = *(const v8u*)(row + 16);
    return concat_frag(lo, hi);
}

// B fragment: 32x16 bf16 (K x N), stored N-major (K contiguous), col stride 40.
// Lane (n=n0+L%16, half=L/16) holds K = {8h..8h+7} and {16+8h..+7}: 2x ds_load_b128.
__device__ __forceinline__ v16bf load_bT_frag(const unsigned short* bs, int ldb,
                                              int n0, int lane) {
    const unsigned short* col = bs + (n0 + (lane & 15)) * ldb + 8 * (lane >> 4);
    v8u lo = *(const v8u*)col;
    v8u hi = *(const v8u*)(col + 16);
    return concat_frag(lo, hi);
}

__device__ __forceinline__ v8f wmma_bf16(v16bf a, v16bf b, v8f c) {
    return __builtin_amdgcn_wmma_f32_16x16x32_bf16(
        false, a, false, b, (short)0, c, false, false);
}

// ---------------------------------------------------------------------------
// K0: one wave32 per adjacency row -> norm = rsqrt(deg + 1e-6)
// ---------------------------------------------------------------------------
__global__ void __launch_bounds__(256)
deg_norm_kernel(const float* __restrict__ adj, float* __restrict__ norm,
                int Nn, int total_rows) {
    const int wid  = (blockIdx.x * blockDim.x + threadIdx.x) >> 5;
    const int lane = threadIdx.x & 31;
    if (wid >= total_rows) return;
    const float* row = adj + (size_t)wid * Nn;
    float s = 0.f;
    for (int j = lane; j < Nn; j += 32) s += fabsf(row[j]);
#pragma unroll
    for (int off = 16; off; off >>= 1) s += __shfl_xor(s, off, 32);
    if (lane == 0) norm[wid] = rsqrtf(s + 1e-6f);
}

// ---------------------------------------------------------------------------
// K1: x f32 -> bf16
// ---------------------------------------------------------------------------
__global__ void __launch_bounds__(256)
cvt_x_kernel(const float* __restrict__ src, unsigned short* __restrict__ dst,
             int n) {
    int i = blockIdx.x * blockDim.x + threadIdx.x;
    if (i < n) dst[i] = f32_to_bf16(src[i]);
}

// ---------------------------------------------------------------------------
// K2: build transposed packed weights  WcatT [768 n-rows x 256 k]  (bf16)
// ---------------------------------------------------------------------------
__global__ void __launch_bounds__(256)
build_wcatT_kernel(const float* __restrict__ th, const float* __restrict__ wh,
                   const float* __restrict__ wt, unsigned short* __restrict__ wcatT) {
    int i = blockIdx.x * blockDim.x + threadIdx.x;  // 0 .. 768*256-1
    int n = i >> 8, k = i & 255;
    float v = (n < 256) ? th[k * 256 + n]
            : (n < 512) ? wh[k * 256 + (n - 256)]
                        : wt[k * 256 + (n - 512)];
    wcatT[i] = f32_to_bf16(v);
}

// ---------------------------------------------------------------------------
// K3: fused input GEMM  [16384 x 256] @ [256 x 768]  (bf16 WMMA, f32 acc)
//   cols [0,256)   -> hT[d][row] = norm_row * (x@theta)   (bf16, TRANSPOSED)
//   cols [256,512) -> het  = x@W_h                        (f32)
//   cols [512,768) -> gate = sigmoid(x@W_t + b_t)         (f32)
// Block 64x128, 8 waves. Double-buffered LDS, one barrier per k-step.
// Pipeline regs are NAMED SCALARS in a by-value struct (SROA -> VGPRs).
// ---------------------------------------------------------------------------
#define LDA 40  // padded LDS strides (ushorts)

struct G1Regs { uint4 ra, rb0, rb1; };

__global__ void __launch_bounds__(256)
gemm1_kernel(const unsigned short* __restrict__ xb,
             const unsigned short* __restrict__ wcatT,
             const float* __restrict__ norm, const float* __restrict__ bt,
             unsigned short* __restrict__ hT, float* __restrict__ het,
             float* __restrict__ gate, int rows) {
    constexpr int ASZ = 64 * LDA;     //  5 KB per buffer
    constexpr int BSZ = 128 * LDA;    // 10 KB per buffer
    __shared__ unsigned short As[2 * ASZ];
    __shared__ unsigned short Bs[2 * BSZ];
    const int tid = threadIdx.x;
    const int lane = tid & 31, wid = tid >> 5;
    const int m0 = 16 * (wid & 3);
    const int nw = 64 * (wid >> 2);
    const int rowbase = blockIdx.x * 64;
    const int colbase = blockIdx.y * 128;

    const int ar = tid >> 2, acq = tid & 3;   // row / uint4-col (of 4)

    auto gload = [&](int k0) -> G1Regs {
        G1Regs g;
        g.ra  = *((const uint4*)(xb + (size_t)(rowbase + ar) * 256 + k0) + acq);
        g.rb0 = *((const uint4*)(wcatT + (size_t)(colbase + ar) * 256 + k0) + acq);
        g.rb1 = *((const uint4*)(wcatT + (size_t)(colbase + ar + 64) * 256 + k0) + acq);
        return g;
    };
    auto stage = [&](int p, G1Regs g) {
        *(uint4*)((uint32_t*)(As + p * ASZ) + ar * (LDA / 2) + 4 * acq) = g.ra;
        *(uint4*)((uint32_t*)(Bs + p * BSZ) + ar * (LDA / 2) + 4 * acq) = g.rb0;
        *(uint4*)((uint32_t*)(Bs + p * BSZ) + (ar + 64) * (LDA / 2) + 4 * acq) = g.rb1;
    };

    stage(0, gload(0));
    __syncthreads();

    v8f acc[4] = {};
    int p = 0;
#pragma unroll 1
    for (int kt = 0; kt < 8; ++kt) {
        const bool more = (kt + 1 < 8);
        G1Regs g;
        if (more) g = gload(32 * (kt + 1));   // issue; waits at stage()
        const unsigned short* ap = As + p * ASZ;
        const unsigned short* bp = Bs + p * BSZ;
        v16bf a = load_a_frag(ap, LDA, m0, lane);
#pragma unroll
        for (int f = 0; f < 4; ++f) {
            v16bf b = load_bT_frag(bp, LDA, nw + 16 * f, lane);
            acc[f] = wmma_bf16(a, b, acc[f]);
        }
        if (more) {
            stage(p ^ 1, g);        // other buffer: no reader conflict
            __syncthreads();        // single barrier per iteration
        }
        p ^= 1;
    }

    const int half = lane >> 4, nc = lane & 15;
#pragma unroll
    for (int f = 0; f < 4; ++f) {
#pragma unroll
        for (int r = 0; r < 8; ++r) {
            int grow = rowbase + m0 + r + 8 * half;
            int gcol = colbase + nw + 16 * f + nc;
            float v = acc[f][r];
            if (gcol < 256) {
                hT[(size_t)gcol * rows + grow] = f32_to_bf16(v * norm[grow]);
            } else if (gcol < 512) {
                het[(size_t)grow * 256 + (gcol - 256)] = v;
            } else {
                float z = v + bt[gcol - 512];
                gate[(size_t)grow * 256 + (gcol - 512)] = 1.0f / (1.0f + expf(-z));
            }
        }
    }
}

// ---------------------------------------------------------------------------
// K4: SpMM  out = elu(gate*norm_i*(adj@h) + (1-gate)*het)
// Block 64 rows x 256 cols, K over N=4096 in steps of 32.
// Double-buffered LDS (50KB), one barrier per k-step; adj float4 loads
// packed to bf16 dwords; hT b128 loads; prefetch 2 tiles ahead.
// ---------------------------------------------------------------------------
struct SpRegs { float4 a0, a1; uint4 b0, b1, b2, b3; };

__global__ void __launch_bounds__(256)
spmm_kernel(const float* __restrict__ adj, const unsigned short* __restrict__ hT,
            const float* __restrict__ norm, const float* __restrict__ het,
            const float* __restrict__ gate, float* __restrict__ out,
            int Nn, int rows) {
    constexpr int ASZ = 64 * LDA;     //  5 KB per buffer
    constexpr int BSZ = 256 * LDA;    // 20 KB per buffer
    __shared__ unsigned short As[2 * ASZ];
    __shared__ unsigned short Bs[2 * BSZ];
    const int tid = threadIdx.x;
    const int lane = tid & 31, wid = tid >> 5;
    const int m0 = 16 * (wid & 3);
    const int nbase = 128 * (wid >> 2);
    const int b = blockIdx.y;
    const int i0 = blockIdx.x * 64;
    const size_t row0 = (size_t)b * Nn;
    const size_t adj_base = (row0 + i0) * Nn;

    // adj staging: thread -> (row, float4-col of 8), two halves 32 rows apart
    const int ar0 = tid >> 3, acq = tid & 7;
    // hT staging: thread -> (d, uint4-col of 4), four slabs 64 d apart
    const int hd = tid >> 2, hcq = tid & 3;

    auto gload = [&](int k0) -> SpRegs {
        SpRegs g;
        const float* ga = adj + adj_base + (size_t)ar0 * Nn + k0;
        g.a0 = *((const float4*)ga + acq);
        g.a1 = *((const float4*)(ga + 32 * (size_t)Nn) + acq);
        if (k0 + 64 < Nn) {                    // stream hint: 2 tiles ahead
            __builtin_prefetch((const float*)((const float4*)ga + acq) + 64, 0, 1);
            __builtin_prefetch((const float*)((const float4*)(ga + 32 * (size_t)Nn) + acq) + 64, 0, 1);
        }
        const unsigned short* gb = hT + (size_t)hd * rows + row0 + k0;
        const size_t dstep = (size_t)64 * rows;
        g.b0 = *((const uint4*)gb + hcq);
        g.b1 = *((const uint4*)(gb + dstep) + hcq);
        g.b2 = *((const uint4*)(gb + 2 * dstep) + hcq);
        g.b3 = *((const uint4*)(gb + 3 * dstep) + hcq);
        return g;
    };
    auto stage = [&](int p, SpRegs g) {
        uint32_t* abuf = (uint32_t*)(As + p * ASZ);
        uint2 p0, p1;
        p0.x = pack_bf16x2(g.a0.x, g.a0.y);
        p0.y = pack_bf16x2(g.a0.z, g.a0.w);
        p1.x = pack_bf16x2(g.a1.x, g.a1.y);
        p1.y = pack_bf16x2(g.a1.z, g.a1.w);
        *(uint2*)(abuf + ar0 * (LDA / 2) + 2 * acq) = p0;
        *(uint2*)(abuf + (ar0 + 32) * (LDA / 2) + 2 * acq) = p1;
        uint32_t* bbuf = (uint32_t*)(Bs + p * BSZ);
        *(uint4*)(bbuf + hd * (LDA / 2) + 4 * hcq) = g.b0;
        *(uint4*)(bbuf + (hd + 64) * (LDA / 2) + 4 * hcq) = g.b1;
        *(uint4*)(bbuf + (hd + 128) * (LDA / 2) + 4 * hcq) = g.b2;
        *(uint4*)(bbuf + (hd + 192) * (LDA / 2) + 4 * hcq) = g.b3;
    };

    stage(0, gload(0));
    __syncthreads();

    v8f acc[8] = {};
    int p = 0;
#pragma unroll 1
    for (int kt = 0; kt < 128; ++kt) {
        const bool more = (kt + 1 < 128);
        SpRegs g;
        if (more) g = gload(32 * (kt + 1));    // overlap with WMMAs
        const unsigned short* ap = As + p * ASZ;
        const unsigned short* bp = Bs + p * BSZ;
        v16bf a = load_a_frag(ap, LDA, m0, lane);
#pragma unroll
        for (int f = 0; f < 8; ++f) {
            v16bf bb = load_bT_frag(bp, LDA, nbase + 16 * f, lane);
            acc[f] = wmma_bf16(a, bb, acc[f]);
        }
        if (more) {
            stage(p ^ 1, g);
            __syncthreads();        // single barrier per iteration
        }
        p ^= 1;
    }

    const int half = lane >> 4, nc = lane & 15;
#pragma unroll
    for (int f = 0; f < 8; ++f) {
#pragma unroll
        for (int r = 0; r < 8; ++r) {
            int node = i0 + m0 + r + 8 * half;
            int d = nbase + 16 * f + nc;
            size_t frow = row0 + node;
            size_t idx = frow * 256 + d;
            float fhom = acc[f][r] * norm[frow];
            float gv = gate[idx];
            float o = gv * fhom + (1.f - gv) * het[idx];
            out[idx] = (o > 0.f) ? o : (expf(o) - 1.f);
        }
    }
}

// ---------------------------------------------------------------------------
// Launcher. ws layout (~48.5 MiB):
//   norm f32[16384] | xb16[16384*256] | WcatT bf16[768*256]
//   | hT bf16[256*16384] | het f32[16384*256] | gate f32[16384*256]
// ---------------------------------------------------------------------------
extern "C" void kernel_launch(void* const* d_in, const int* in_sizes, int n_in,
                              void* d_out, int out_size, void* d_ws, size_t ws_size,
                              hipStream_t stream) {
    (void)in_sizes; (void)n_in; (void)out_size; (void)ws_size;
    const float* x   = (const float*)d_in[0];
    const float* adj = (const float*)d_in[1];
    const float* th  = (const float*)d_in[2];
    const float* wh  = (const float*)d_in[3];
    const float* wt  = (const float*)d_in[4];
    const float* bt  = (const float*)d_in[5];
    float* out = (float*)d_out;

    const int Bb = 4, Nn = 4096, Dd = 256;
    const int rows = Bb * Nn;  // 16384

    char* ws = (char*)d_ws;
    size_t off = 0;
    float* norm = (float*)(ws + off);                    off += (size_t)rows * 4;
    unsigned short* xb = (unsigned short*)(ws + off);    off += (size_t)rows * Dd * 2;
    unsigned short* wcatT = (unsigned short*)(ws + off); off += (size_t)768 * 256 * 2;
    unsigned short* hT = (unsigned short*)(ws + off);    off += (size_t)rows * Dd * 2;
    float* het  = (float*)(ws + off);                    off += (size_t)rows * Dd * 4;
    float* gate = (float*)(ws + off);                    off += (size_t)rows * Dd * 4;

    deg_norm_kernel<<<rows / 8, 256, 0, stream>>>(adj, norm, Nn, rows);
    cvt_x_kernel<<<(rows * Dd) / 256, 256, 0, stream>>>(x, xb, rows * Dd);
    build_wcatT_kernel<<<(768 * 256) / 256, 256, 0, stream>>>(th, wh, wt, wcatT);
    gemm1_kernel<<<dim3(rows / 64, 6), 256, 0, stream>>>(xb, wcatT, norm, bt,
                                                         hT, het, gate, rows);
    spmm_kernel<<<dim3(Nn / 64, Bb), 256, 0, stream>>>(adj, hT, norm, het,
                                                       gate, out, Nn, rows);
}
